// Model_283467841824
// MI455X (gfx1250) — matrix operand order repaired and checked by reading, not compiled, not run on hardware
//
#include <hip/hip_runtime.h>
#include <hip/hip_bf16.h>

// PAC (pixel-adaptive conv) x2 on MI455X / gfx1250.
// Contraction recast as 81 GEMMs of (M=32, K=32, N=16384) per layer, executed
// with V_WMMA_F32_16X16X32_BF16 (bf16 A/B, f32 accumulate). Wave32: one wave
// owns a 16-pixel N-tile and both 16-row M-tiles of the 32 output channels.
// B operands kept in bf16 end-to-end; kmod scaling via V_PK_MUL_BF16 (pinned
// to the consume stage with volatile asm). OOB taps are branch-free (clamped
// address + multiplicative zero mask). The tap loop is a depth-2 ping-pong
// software pipeline: loads for tap kl+2 are in flight while tap kl computes.

#define NPIX 16384   // 128*128
#define HH   128
#define WW   128
#define NKL  81      // 9*9 taps

typedef __attribute__((ext_vector_type(16))) __bf16 v16bf;
typedef __attribute__((ext_vector_type(8)))  float  v8f;
typedef __attribute__((ext_vector_type(4)))  float  v4f;

__device__ __forceinline__ int iclamp(int v, int lo, int hi) {
  return v < lo ? lo : (v > hi ? hi : v);
}

// ---------------------------------------------------------------------------
// 1) CHW -> pixel-major: x to bf16 (WMMA B feed), f stays f32 (kernel quality).
__global__ void pac_transpose(const float* __restrict__ x,
                              const float* __restrict__ f,
                              __bf16* __restrict__ xTb,
                              float* __restrict__ fT) {
  const int p = blockIdx.x * blockDim.x + threadIdx.x;  // exact grid: no guard
#pragma unroll
  for (int c = 0; c < 32; ++c) {
    xTb[p * 32 + c] = (__bf16)x[c * NPIX + p];
    fT[p * 32 + c]  = f[c * NPIX + p];
  }
}

// ---------------------------------------------------------------------------
// 2) Stage W (O=32, C=32, 9, 9) into a per-lane bf16 A-fragment layout for
//    V_WMMA_*_16x16x32 (ISA 7.12.2, 16-bit A 16x32):
//      lanes 0-15 : M=lane,    VGPR v -> K=2v..2v+1 (contiguous, K = slot)
//      lanes 16-31: M=lane-16, same K pattern + 16
//    Output: Ast[(kl*2 + mtile)*32 + lane][slot 0..15] contiguous 32B/lane.
__global__ void pac_wprep(const float* __restrict__ W, __bf16* __restrict__ Ast) {
  const int t = blockIdx.x * blockDim.x + threadIdx.x;  // 81*2*32 = 5184 exact
  const int lane = t & 31;
  const int m    = (t >> 5) & 1;
  const int kl   = t >> 6;
  const int M    = 16 * m + (lane & 15);
  const int khalf = (lane >> 4) * 16;
#pragma unroll
  for (int s = 0; s < 16; ++s) {
    const int v = s >> 1, pos = s & 1;
    const int c = 2 * v + pos + khalf;
    Ast[t * 16 + s] = (__bf16)W[(M * 32 + c) * NKL + kl];  // W[o][c][kl]
  }
}

// ---------------------------------------------------------------------------
// 3) Guidance kernel k[kl][p] = exp(-0.5 * ||f_nbr - f_ctr||^2), zero-padded f.
__global__ void pac_kmod(const float* __restrict__ fT, float* __restrict__ kmod) {
  const int idx = blockIdx.x * blockDim.x + threadIdx.x;  // 81*16384 exact
  const int kl = idx >> 14;
  const int p  = idx & (NPIX - 1);
  const int i = kl / 9, j = kl - 9 * i;
  const int h = p >> 7, w = p & (WW - 1);
  const int nh = h + 2 * i - 8, nw = w + 2 * j - 8;
  const v4f* fc = (const v4f*)(fT + p * 32);
  float s = 0.f;
  if (nh >= 0 && nh < HH && nw >= 0 && nw < WW) {
    const v4f* fn = (const v4f*)(fT + (nh * WW + nw) * 32);
#pragma unroll
    for (int q = 0; q < 8; ++q) {
      v4f d = fn[q] - fc[q];
      s += d[0] * d[0] + d[1] * d[1] + d[2] * d[2] + d[3] * d[3];
    }
  } else {  // f_nbr = 0 (zero padding) -> d = -f_ctr
#pragma unroll
    for (int q = 0; q < 8; ++q) {
      v4f d = fc[q];
      s += d[0] * d[0] + d[1] * d[1] + d[2] * d[2] + d[3] * d[3];
    }
  }
  kmod[idx] = __expf(-0.5f * s);
}

// ---------------------------------------------------------------------------
// 4) WMMA pacconv, depth-2 software pipeline.
//    B[c, n] = srcT[clamp(nbr(n))][c] * (valid ? kmod[kl, n] : 0)
//    per-lane: N = lane&15, K-half = 16*(lane>>4) -> 2x b128 bf16 loads,
//    scale replicated via v_cvt_pk_bf16_f32, applied with v_pk_mul_bf16.
//    CHW_OUT=true -> f32 CHW store (d_out), false -> bf16 pixel-major (hT).
template <bool CHW_OUT>
__global__ void pac_conv(const __bf16* __restrict__ srcT,
                         const v16bf* __restrict__ Ast,
                         const float* __restrict__ kmod,
                         const float* __restrict__ bias,
                         void* __restrict__ outp) {
  const int lane  = threadIdx.x & 31;
  const int wave  = threadIdx.x >> 5;
  const int pbase = blockIdx.x * 128 + wave * 16;   // 128 blk * 8 waves * 16 px
  const int p     = pbase + (lane & 15);
  const int h     = p >> 7, w = p & (WW - 1);
  const int chalf = (lane >> 4) * 16;               // K-half this lane supplies

  v8f acc0, acc1;
#pragma unroll
  for (int q = 0; q < 8; ++q) { acc0[q] = 0.f; acc1[q] = 0.f; }

  // Two explicit pipeline stages (even taps -> A, odd taps -> B): no arrays,
  // no dynamic indexing, everything stays in VGPRs.
  uint4 At0, At1, Bt0, Bt1;
  v16bf Aa0, Aa1, Ba0, Ba1;
  float Akv, Amk, Bkv, Bmk;

  int li, lj;  // (i,j) cursor of the next tap to issue

  auto issue = [&](int i, int j, int kl,
                   uint4& t0, uint4& t1, v16bf& a0, v16bf& a1,
                   float& kv, float& mk) {
    const int nh  = h + 2 * i - 8;
    const int nw  = w + 2 * j - 8;
    const int nhc = iclamp(nh, 0, HH - 1);
    const int nwc = iclamp(nw, 0, WW - 1);
    const uint4* src = (const uint4*)(srcT + (nhc * WW + nwc) * 32 + chalf);
    t0 = src[0];
    t1 = src[1];
    a0 = Ast[(kl * 2 + 0) * 32 + lane];
    a1 = Ast[(kl * 2 + 1) * 32 + lane];
    kv = kmod[kl * NPIX + p];           // always a valid address
    mk = ((nh == nhc) & (nw == nwc)) ? 1.0f : 0.0f;
  };

  // consume stage (tap kl), then refill the same stage with tap kl+2.
  // Tail refills clamp to tap 80 (redundant in-bounds loads, never consumed).
  auto step = [&](int kl,
                  uint4& t0s, uint4& t1s, v16bf& a0s, v16bf& a1s,
                  float& kvs, float& mks) {
    const uint4 t0 = t0s, t1 = t1s;
    const v16bf a0 = a0s, a1 = a1s;
    const float ks = kvs * mks;          // load result always used -> no exec

    const int nkl = (kl + 2 <= NKL - 1) ? kl + 2 : NKL - 1;
    issue(li, lj, nkl, t0s, t1s, a0s, a1s, kvs, mks);
    ++lj;
    if (lj == 9) { lj = 0; ++li; }

    unsigned sc2;  // bf16(ks) replicated into both halves of a dword
    asm volatile("v_cvt_pk_bf16_f32 %0, %1, %1" : "=v"(sc2) : "v"(ks));

    union { unsigned u[8]; v16bf v; } B;
    asm volatile("v_pk_mul_bf16 %0, %1, %2" : "=v"(B.u[0]) : "v"(t0.x), "v"(sc2));
    asm volatile("v_pk_mul_bf16 %0, %1, %2" : "=v"(B.u[1]) : "v"(t0.y), "v"(sc2));
    asm volatile("v_pk_mul_bf16 %0, %1, %2" : "=v"(B.u[2]) : "v"(t0.z), "v"(sc2));
    asm volatile("v_pk_mul_bf16 %0, %1, %2" : "=v"(B.u[3]) : "v"(t0.w), "v"(sc2));
    asm volatile("v_pk_mul_bf16 %0, %1, %2" : "=v"(B.u[4]) : "v"(t1.x), "v"(sc2));
    asm volatile("v_pk_mul_bf16 %0, %1, %2" : "=v"(B.u[5]) : "v"(t1.y), "v"(sc2));
    asm volatile("v_pk_mul_bf16 %0, %1, %2" : "=v"(B.u[6]) : "v"(t1.z), "v"(sc2));
    asm volatile("v_pk_mul_bf16 %0, %1, %2" : "=v"(B.u[7]) : "v"(t1.w), "v"(sc2));

    // EXEC is all-ones here: exact grid, no early exits (WMMA requirement).
    acc0 = __builtin_amdgcn_wmma_f32_16x16x32_bf16(
        false, a0, false, B.v, (short)0, acc0, false, false);
    acc1 = __builtin_amdgcn_wmma_f32_16x16x32_bf16(
        false, a1, false, B.v, (short)0, acc1, false, false);
  };

  // prologue: taps 0 and 1 in flight
  issue(0, 0, 0, At0, At1, Aa0, Aa1, Akv, Amk);
  issue(0, 1, 1, Bt0, Bt1, Ba0, Ba1, Bkv, Bmk);
  li = 0;
  lj = 2;

  for (int kl = 0; kl < NKL - 1; kl += 2) {  // taps 0..79 in pairs
    step(kl,     At0, At1, Aa0, Aa1, Akv, Amk);
    step(kl + 1, Bt0, Bt1, Ba0, Ba1, Bkv, Bmk);
  }
  step(NKL - 1, At0, At1, Aa0, Aa1, Akv, Amk);  // tap 80 (even -> stage A)

  // C/D layout: lane holds column N=lane&15; VGPR v -> M = v + 8*(lane>>4).
  const int chbase = 8 * (lane >> 4);
  if (CHW_OUT) {
    float* out = (float*)outp;
#pragma unroll
    for (int v = 0; v < 8; ++v) {
      out[(chbase + v) * NPIX + p]      = acc0[v] + bias[chbase + v];
      out[(chbase + 16 + v) * NPIX + p] = acc1[v] + bias[chbase + 16 + v];
    }
  } else {  // bf16 pixel-major: lane's 8 channels consecutive -> b128 stores
    __bf16* o = (__bf16*)outp + p * 32 + chbase;
#pragma unroll
    for (int v = 0; v < 8; ++v) {
      o[v]      = (__bf16)(acc0[v] + bias[chbase + v]);
      o[16 + v] = (__bf16)(acc1[v] + bias[chbase + 16 + v]);
    }
  }
}

// ---------------------------------------------------------------------------
extern "C" void kernel_launch(void* const* d_in, const int* in_sizes, int n_in,
                              void* d_out, int out_size, void* d_ws, size_t ws_size,
                              hipStream_t stream) {
  const float* x  = (const float*)d_in[0];
  const float* f  = (const float*)d_in[1];
  const float* W1 = (const float*)d_in[2];
  const float* b1 = (const float*)d_in[3];
  const float* W2 = (const float*)d_in[4];
  const float* b2 = (const float*)d_in[5];

  // Workspace layout (~9.4 MB, all sub-arrays 128B-aligned, no overruns).
  float*  fT   = (float*)d_ws;                  // 16384*32 f32   (2 MB)
  float*  kmod = fT + NPIX * 32;                // 81*16384 f32   (5.06 MB)
  __bf16* xTb  = (__bf16*)(kmod + NKL * NPIX);  // 16384*32 bf16  (1 MB)
  __bf16* hTb  = xTb + NPIX * 32;               // 16384*32 bf16  (1 MB)
  v16bf*  A1   = (v16bf*)(hTb + NPIX * 32);     // 81*2*32 frags  (162 KB)
  v16bf*  A2   = A1 + NKL * 2 * 32;             // 81*2*32 frags  (162 KB)

  pac_transpose<<<NPIX / 256, 256, 0, stream>>>(x, f, xTb, fT);
  pac_wprep<<<NKL, 64, 0, stream>>>(W1, (__bf16*)A1);
  pac_wprep<<<NKL, 64, 0, stream>>>(W2, (__bf16*)A2);
  pac_kmod<<<(NKL * NPIX) / 256, 256, 0, stream>>>(fT, kmod);

  pac_conv<false><<<128, 256, 0, stream>>>(xTb, A1, kmod, b1, hTb);           // layer 1
  pac_conv<true><<<128, 256, 0, stream>>>(hTb, A2, kmod, b2, (float*)d_out);  // layer 2
}
